// LlamaDecoderBlock_89893665506094
// MI455X (gfx1250) — matrix operand order repaired and checked by reading, not compiled, not run on hardware
//
#include <hip/hip_runtime.h>
#include <hip/hip_bf16.h>
#include <math.h>

// ---------------------------------------------------------------- types
typedef __bf16 bf16;
typedef __attribute__((ext_vector_type(16))) __bf16 bf16x16;
typedef __attribute__((ext_vector_type(8)))  float  f32x8;

#define D_MODEL 2048
#define N_HEADS 16
#define HEAD_DIM 128
#define D_FF 8192
#define BB 2
#define TT 2048
#define NTOK (BB*TT)       // 4096 tokens
#define EPSV 1e-6f

union FragU { bf16x16 v; uint4 u[2]; bf16 e[16]; };

// Build a 16x32 bf16 WMMA fragment from two 16-byte chunks.
__device__ __forceinline__ bf16x16 make_frag(const bf16* p0, const bf16* p1) {
  FragU f;
  f.u[0] = *reinterpret_cast<const uint4*>(p0);
  f.u[1] = *reinterpret_cast<const uint4*>(p1);
  return f.v;
}

// Issue one async 16B global->LDS copy (per lane), tracked by ASYNCcnt.
__device__ __forceinline__ void async_b128(const bf16* gptr, bf16* lptr) {
  uint32_t loff = (uint32_t)(uintptr_t)lptr;           // low 32b = LDS offset
  uint64_t gaddr = (uint64_t)(uintptr_t)gptr;
  asm volatile("global_load_async_to_lds_b128 %0, %1, off"
               :: "v"(loff), "v"(gaddr) : "memory");
}

// ---------------------------------------------------------------- transpose cast
// W fp32 [K][N] -> Wt bf16 [N][K], LDS-tiled 32x32 for coalescing both sides.
__global__ void transpose_cast_bf16(const float* __restrict__ W,
                                    bf16* __restrict__ Wt, int K, int N) {
  __shared__ bf16 t[32][33];
  const int k0 = blockIdx.y * 32, n0 = blockIdx.x * 32;
  const int r = threadIdx.x >> 5, c = threadIdx.x & 31;
  #pragma unroll
  for (int i = 0; i < 4; ++i)
    t[r + 8 * i][c] = (bf16)W[(size_t)(k0 + r + 8 * i) * N + n0 + c];
  __syncthreads();
  #pragma unroll
  for (int i = 0; i < 4; ++i)
    Wt[(size_t)(n0 + r + 8 * i) * K + k0 + c] = t[c][r + 8 * i];
}

// ---------------------------------------------------------------- rmsnorm
__global__ void rmsnorm_bf16(const float* __restrict__ x,
                             const float* __restrict__ w,
                             bf16* __restrict__ out) {
  int row = blockIdx.x;
  const float* xr = x + (size_t)row * D_MODEL;
  __shared__ float red[256];
  float s = 0.f;
  for (int c = threadIdx.x; c < D_MODEL; c += 256) { float v = xr[c]; s += v * v; }
  red[threadIdx.x] = s;
  __syncthreads();
  for (int o = 128; o > 0; o >>= 1) {
    if ((int)threadIdx.x < o) red[threadIdx.x] += red[threadIdx.x + o];
    __syncthreads();
  }
  float inv = rsqrtf(red[0] / (float)D_MODEL + EPSV);
  for (int c = threadIdx.x; c < D_MODEL; c += 256)
    out[(size_t)row * D_MODEL + c] = (bf16)(xr[c] * inv * w[c]);
}

// ---------------------------------------------------------------- GEMM
// C[M,N] = A[M,K](bf16, row-major) * Bt[N,K](bf16, pre-transposed) [+bias][+resid]
// Block tile 256x128x32, wave tile 64x64 (4x4 WMMA tiles): 16 WMMAs per
// 16 ds_load_b128 per K-step. Double-buffered LDS filled by async
// global->LDS loads; per-wave batch = 6 async instrs -> s_wait_asynccnt 0x6.
template<bool BIAS, bool RESID>
__global__ __launch_bounds__(256)
void gemm_bf16(const bf16* __restrict__ A, const bf16* __restrict__ Bt,
               const float* __restrict__ bias, const float* __restrict__ resid,
               float* __restrict__ C, int K, int N) {
  __shared__ __attribute__((aligned(16))) bf16 lA[2][256 * 32];
  __shared__ __attribute__((aligned(16))) bf16 lB[2][128 * 32];
  const int tid = threadIdx.x;
  const int lane = tid & 31, wid = tid >> 5;
  const int wm = wid & 3, wn = wid >> 2;          // 4x2 wave grid
  const int bm = blockIdx.y * 256, bn = blockIdx.x * 128;
  const int hf = lane >> 4, l16 = lane & 15;

  f32x8 acc[4][4] = {};

  // stage one 256x32 A tile (4 loads) + 128x32 Bt tile (2 loads) per thread
  auto stage = [&](int kb, int buf) {
    #pragma unroll
    for (int i = 0; i < 4; ++i) {
      int idx = tid * 8 + i * 2048;               // element index within A tile
      int r = idx >> 5, c = idx & 31;
      async_b128(A + (size_t)(bm + r) * K + kb + c, &lA[buf][idx]);
    }
    #pragma unroll
    for (int i = 0; i < 2; ++i) {
      int idx = tid * 8 + i * 2048;               // element index within B tile
      int r = idx >> 5, c = idx & 31;
      async_b128(Bt + (size_t)(bn + r) * K + kb + c, &lB[buf][idx]);
    }
  };

  const int nkb = K >> 5;
  stage(0, 0);
  for (int kb = 0; kb < nkb; ++kb) {
    const int cur = kb & 1;
    if (kb + 1 < nkb) {
      stage((kb + 1) * 32, cur ^ 1);
      asm volatile("s_wait_asynccnt 0x6" ::: "memory");  // batch kb landed
    } else {
      asm volatile("s_wait_asynccnt 0x0" ::: "memory");
    }
    __syncthreads();

    bf16x16 afrag[4];
    #pragma unroll
    for (int mt = 0; mt < 4; ++mt) {
      const bf16* p = &lA[cur][(wm * 64 + mt * 16 + l16) * 32 + 8 * hf];
      afrag[mt] = make_frag(p, p + 16);      // ISA 16-bit A layout
    }
    #pragma unroll
    for (int nt = 0; nt < 4; ++nt) {
      const bf16* p = &lB[cur][(wn * 64 + nt * 16 + l16) * 32 + 8 * hf];
      bf16x16 bfrag = make_frag(p, p + 16);  // lane = column, ISA 16-bit K layout
      #pragma unroll
      for (int mt = 0; mt < 4; ++mt)
        acc[mt][nt] = __builtin_amdgcn_wmma_f32_16x16x32_bf16(
            false, afrag[mt], false, bfrag, (short)0, acc[mt][nt], false, false);
    }
    __syncthreads();
  }

  #pragma unroll
  for (int mt = 0; mt < 4; ++mt)
    #pragma unroll
    for (int nt = 0; nt < 4; ++nt)
      #pragma unroll
      for (int j = 0; j < 8; ++j) {
        int r = bm + wm * 64 + mt * 16 + j + 8 * hf;
        int c = bn + wn * 64 + nt * 16 + l16;
        float v = acc[mt][nt][j];
        if constexpr (BIAS)  v += bias[c];
        if constexpr (RESID) v += resid[(size_t)r * N + c];
        C[(size_t)r * N + c] = v;
      }
}

// ---------------------------------------------------------------- gate/up fused
// act = silu(A*Gt^T) * (A*Ut^T), bf16 out; shares A fragments across both
// products (16 WMMAs / K-step). Async double-buffer, batch = 6 -> 0x6.
__global__ __launch_bounds__(256)
void gemm_gateup(const bf16* __restrict__ A, const bf16* __restrict__ Gt,
                 const bf16* __restrict__ Ut, bf16* __restrict__ act,
                 int K, int N) {
  __shared__ __attribute__((aligned(16))) bf16 lA[2][128 * 32];
  __shared__ __attribute__((aligned(16))) bf16 lG[2][128 * 32];
  __shared__ __attribute__((aligned(16))) bf16 lU[2][128 * 32];
  const int tid = threadIdx.x;
  const int lane = tid & 31, wid = tid >> 5;
  const int wm = wid & 3, wn = wid >> 2;
  const int bm = blockIdx.y * 128, bn = blockIdx.x * 128;
  const int hf = lane >> 4, l16 = lane & 15;

  f32x8 accg[2][4] = {};
  f32x8 accu[2][4] = {};

  auto stage = [&](int kb, int buf) {
    #pragma unroll
    for (int i = 0; i < 2; ++i) {
      int idx = tid * 8 + i * 2048;
      int r = idx >> 5, c = idx & 31;
      async_b128(A  + (size_t)(bm + r) * K + kb + c, &lA[buf][idx]);
      async_b128(Gt + (size_t)(bn + r) * K + kb + c, &lG[buf][idx]);
      async_b128(Ut + (size_t)(bn + r) * K + kb + c, &lU[buf][idx]);
    }
  };

  const int nkb = K >> 5;
  stage(0, 0);
  for (int kb = 0; kb < nkb; ++kb) {
    const int cur = kb & 1;
    if (kb + 1 < nkb) {
      stage((kb + 1) * 32, cur ^ 1);
      asm volatile("s_wait_asynccnt 0x6" ::: "memory");
    } else {
      asm volatile("s_wait_asynccnt 0x0" ::: "memory");
    }
    __syncthreads();

    bf16x16 afrag[2];
    #pragma unroll
    for (int mt = 0; mt < 2; ++mt) {
      const bf16* p = &lA[cur][(wm * 32 + mt * 16 + l16) * 32 + 8 * hf];
      afrag[mt] = make_frag(p, p + 16);
    }
    #pragma unroll
    for (int nt = 0; nt < 4; ++nt) {
      const bf16* pg = &lG[cur][(wn * 64 + nt * 16 + l16) * 32 + 8 * hf];
      const bf16* pu = &lU[cur][(wn * 64 + nt * 16 + l16) * 32 + 8 * hf];
      bf16x16 gfrag = make_frag(pg, pg + 16);
      bf16x16 ufrag = make_frag(pu, pu + 16);
      #pragma unroll
      for (int mt = 0; mt < 2; ++mt) {
        accg[mt][nt] = __builtin_amdgcn_wmma_f32_16x16x32_bf16(
            false, afrag[mt], false, gfrag, (short)0, accg[mt][nt], false, false);
        accu[mt][nt] = __builtin_amdgcn_wmma_f32_16x16x32_bf16(
            false, afrag[mt], false, ufrag, (short)0, accu[mt][nt], false, false);
      }
    }
    __syncthreads();
  }

  #pragma unroll
  for (int mt = 0; mt < 2; ++mt)
    #pragma unroll
    for (int nt = 0; nt < 4; ++nt)
      #pragma unroll
      for (int j = 0; j < 8; ++j) {
        int r = bm + wm * 32 + mt * 16 + j + 8 * hf;
        int c = bn + wn * 64 + nt * 16 + l16;
        float g = accg[mt][nt][j];
        float u = accu[mt][nt][j];
        float silu = g / (1.f + __expf(-g));
        act[(size_t)r * N + c] = (bf16)(silu * u);
      }
}

// ---------------------------------------------------------------- RoPE on q,k
// qkv fp32 [B,T,3D] -> q_bf,k_bf [B,H,T,HD]  (one thread per rotation pair)
__global__ void rope_qk(const float* __restrict__ qkv,
                        bf16* __restrict__ qb, bf16* __restrict__ kbuf) {
  int i = blockIdx.x * 256 + threadIdx.x;
  if (i >= BB * TT * N_HEADS * (HEAD_DIM / 2)) return;
  int p = i & 63;
  int h = (i >> 6) & 15;
  int t = (i >> 10) & (TT - 1);
  int b = i >> 21;
  float inv = __powf(10000.f, -(float)(2 * p) / (float)HEAD_DIM);
  float ang = (float)t * inv;
  float c = __cosf(ang), s = __sinf(ang);
  size_t base = ((size_t)(b * TT + t)) * (3 * D_MODEL) + h * HEAD_DIM + 2 * p;
  float q1 = qkv[base],            q2 = qkv[base + 1];
  float k1 = qkv[base + D_MODEL],  k2 = qkv[base + D_MODEL + 1];
  size_t ob = ((size_t)((b * N_HEADS + h) * TT + t)) * HEAD_DIM + 2 * p;
  qb[ob]     = (bf16)(q1 * c - q2 * s);
  qb[ob + 1] = (bf16)(q1 * s + q2 * c);
  kbuf[ob]     = (bf16)(k1 * c - k2 * s);
  kbuf[ob + 1] = (bf16)(k1 * s + k2 * c);
}

// V cast + transpose: qkv fp32 -> v_t bf16 [B,H,HD,T]
__global__ void vcast_t(const float* __restrict__ qkv, bf16* __restrict__ vt) {
  int i = blockIdx.x * 256 + threadIdx.x;
  if (i >= BB * TT * D_MODEL) return;
  int d = i & 127;
  int h = (i >> 7) & 15;
  int t = (i >> 11) & (TT - 1);
  int b = i >> 22;
  float v = qkv[((size_t)(b * TT + t)) * (3 * D_MODEL) + 2 * D_MODEL + h * HEAD_DIM + d];
  vt[((size_t)((b * N_HEADS + h) * HEAD_DIM + d)) * TT + t] = (bf16)v;
}

// ---------------------------------------------------------------- flash attention
// Q,K [B,H,T,HD] bf16; Vt [B,H,HD,T] bf16 -> O [B,T,D] bf16 (head-concat).
// Block: 128 query rows (8 waves x 16 rows), keys streamed in blocks of 32.
__global__ __launch_bounds__(256)
void flash_attn(const bf16* __restrict__ Q, const bf16* __restrict__ Km,
                const bf16* __restrict__ Vt, bf16* __restrict__ O) {
  __shared__ __attribute__((aligned(16))) bf16 lP[8][16 * 32];  // wave-private P
  const int lane = threadIdx.x & 31, wid = threadIdx.x >> 5;
  const int hf = lane >> 4, l16 = lane & 15;
  const int bh = blockIdx.y;            // b*H + h
  const int h = bh & (N_HEADS - 1), b = bh >> 4;
  const int q0 = blockIdx.x * 128 + wid * 16;

  const bf16* qp = Q  + (size_t)bh * TT * HEAD_DIM;
  const bf16* kp = Km + (size_t)bh * TT * HEAD_DIM;
  const bf16* vp = Vt + (size_t)bh * HEAD_DIM * TT;

  // Q fragments for all 4 contraction steps, held in registers.
  bf16x16 qf[4];
  #pragma unroll
  for (int ks = 0; ks < 4; ++ks) {
    const bf16* p = qp + (size_t)(q0 + l16) * HEAD_DIM + ks * 32 + 8 * hf;
    qf[ks] = make_frag(p, p + 16);
  }

  f32x8 accO[8] = {};
  float m_run[8], l_run[8];
  #pragma unroll
  for (int j = 0; j < 8; ++j) { m_run[j] = -INFINITY; l_run[j] = 0.f; }

  const float scale = 0.08838834764831845f;  // 1/sqrt(HEAD_DIM)
  const int nkb = (q0 + 16 + 31) >> 5;       // causal trip count for this wave

  for (int kb = 0; kb < nkb; ++kb) {
    const int key0 = kb * 32;
    // S = Q K^T : K fragments load straight from global (HD contiguous per key)
    f32x8 s0 = {}, s1 = {};
    #pragma unroll
    for (int ks = 0; ks < 4; ++ks) {
      const bf16* p0 = kp + (size_t)(key0 + l16) * HEAD_DIM + ks * 32 + 8 * hf;
      const bf16* p1 = kp + (size_t)(key0 + 16 + l16) * HEAD_DIM + ks * 32 + 8 * hf;
      bf16x16 k0 = make_frag(p0, p0 + 16);
      bf16x16 k1 = make_frag(p1, p1 + 16);
      s0 = __builtin_amdgcn_wmma_f32_16x16x32_bf16(false, qf[ks], false, k0,
                                                   (short)0, s0, false, false);
      s1 = __builtin_amdgcn_wmma_f32_16x16x32_bf16(false, qf[ks], false, k1,
                                                   (short)0, s1, false, false);
    }

    // causal mask + online softmax (rows live in 16-lane half-groups)
    float p0v[8], p1v[8], alpha[8];
    #pragma unroll
    for (int j = 0; j < 8; ++j) {
      int row = q0 + j + 8 * hf;
      int c0 = key0 + l16, c1 = key0 + 16 + l16;
      float v0 = (c0 <= row) ? s0[j] * scale : -INFINITY;
      float v1 = (c1 <= row) ? s1[j] * scale : -INFINITY;
      float t = fmaxf(v0, v1);
      t = fmaxf(t, __shfl_xor(t, 1, 32));
      t = fmaxf(t, __shfl_xor(t, 2, 32));
      t = fmaxf(t, __shfl_xor(t, 4, 32));
      t = fmaxf(t, __shfl_xor(t, 8, 32));
      float mn = fmaxf(m_run[j], t);
      float a  = __expf(m_run[j] - mn);
      float e0 = __expf(v0 - mn);
      float e1 = __expf(v1 - mn);
      float rs = e0 + e1;
      rs += __shfl_xor(rs, 1, 32);
      rs += __shfl_xor(rs, 2, 32);
      rs += __shfl_xor(rs, 4, 32);
      rs += __shfl_xor(rs, 8, 32);
      l_run[j] = l_run[j] * a + rs;
      m_run[j] = mn;
      alpha[j] = a;
      p0v[j] = e0; p1v[j] = e1;
    }
    #pragma unroll
    for (int nt = 0; nt < 8; ++nt)
      #pragma unroll
      for (int j = 0; j < 8; ++j) accO[nt][j] *= alpha[j];

    // P: C-layout -> A-layout via wave-private LDS (wave32 lockstep + dscnt)
    #pragma unroll
    for (int j = 0; j < 8; ++j) {
      lP[wid][(j + 8 * hf) * 32 + l16]      = (bf16)p0v[j];
      lP[wid][(j + 8 * hf) * 32 + 16 + l16] = (bf16)p1v[j];
    }
    asm volatile("s_wait_dscnt 0" ::: "memory");
    const bf16* pp = &lP[wid][l16 * 32 + 8 * hf];
    bf16x16 pf = make_frag(pp, pp + 16);

    // O += P V : V fragments load straight from global (T contiguous per dim)
    #pragma unroll
    for (int nt = 0; nt < 8; ++nt) {
      const bf16* vptr = vp + (size_t)(nt * 16 + l16) * TT + key0 + 8 * hf;
      bf16x16 vf = make_frag(vptr, vptr + 16);
      accO[nt] = __builtin_amdgcn_wmma_f32_16x16x32_bf16(false, pf, false, vf,
                                                         (short)0, accO[nt], false, false);
    }
  }

  #pragma unroll
  for (int j = 0; j < 8; ++j) {
    float invl = 1.f / l_run[j];
    int row = q0 + j + 8 * hf;
    size_t base = ((size_t)(b * TT + row)) * D_MODEL + h * HEAD_DIM;
    #pragma unroll
    for (int nt = 0; nt < 8; ++nt)
      O[base + nt * 16 + l16] = (bf16)(accO[nt][j] * invl);
  }
}

// ---------------------------------------------------------------- launcher
extern "C" void kernel_launch(void* const* d_in, const int* in_sizes, int n_in,
                              void* d_out, int out_size, void* d_ws, size_t ws_size,
                              hipStream_t stream) {
  (void)in_sizes; (void)n_in; (void)out_size; (void)ws_size;
  const float* x       = (const float*)d_in[0];
  // d_in[1] = mask (implied by causal masking in flash_attn)
  const float* norm1_w = (const float*)d_in[2];
  const float* qkv_w   = (const float*)d_in[3];
  const float* qkv_b   = (const float*)d_in[4];
  const float* out_w   = (const float*)d_in[5];
  const float* out_b   = (const float*)d_in[6];
  const float* norm2_w = (const float*)d_in[7];
  const float* w_gate  = (const float*)d_in[8];
  const float* w_up    = (const float*)d_in[9];
  const float* w_down  = (const float*)d_in[10];
  float* out = (float*)d_out;

  char* wsb = (char*)d_ws;
  size_t off = 0;
  auto alloc = [&](size_t bytes) -> char* {
    char* p = wsb + off;
    off += (bytes + 255) & ~(size_t)255;
    return p;
  };
  bf16*  wqkv  = (bf16*)alloc((size_t)D_MODEL * 3 * D_MODEL * 2);  // [6144][2048]
  bf16*  wout  = (bf16*)alloc((size_t)D_MODEL * D_MODEL * 2);      // [2048][2048]
  bf16*  wgate = (bf16*)alloc((size_t)D_MODEL * D_FF * 2);         // [8192][2048]
  bf16*  wup   = (bf16*)alloc((size_t)D_MODEL * D_FF * 2);         // [8192][2048]
  bf16*  wdown = (bf16*)alloc((size_t)D_FF * D_MODEL * 2);         // [2048][8192]
  bf16*  h1    = (bf16*)alloc((size_t)NTOK * D_MODEL * 2);
  float* qkvf  = (float*)alloc((size_t)NTOK * 3 * D_MODEL * 4);
  bf16*  qb    = (bf16*)alloc((size_t)NTOK * D_MODEL * 2);
  bf16*  kbuf  = (bf16*)alloc((size_t)NTOK * D_MODEL * 2);
  bf16*  vt    = (bf16*)alloc((size_t)NTOK * D_MODEL * 2);
  bf16*  attnb = (bf16*)alloc((size_t)NTOK * D_MODEL * 2);
  bf16*  h2    = (bf16*)alloc((size_t)NTOK * D_MODEL * 2);
  bf16*  actb  = (bf16*)alloc((size_t)NTOK * D_FF * 2);

  // 1) weight transpose+cast (once per call; removes per-tile LDS transposes)
  transpose_cast_bf16<<<dim3(3 * D_MODEL / 32, D_MODEL / 32), 256, 0, stream>>>(
      qkv_w, wqkv, D_MODEL, 3 * D_MODEL);
  transpose_cast_bf16<<<dim3(D_MODEL / 32, D_MODEL / 32), 256, 0, stream>>>(
      out_w, wout, D_MODEL, D_MODEL);
  transpose_cast_bf16<<<dim3(D_FF / 32, D_MODEL / 32), 256, 0, stream>>>(
      w_gate, wgate, D_MODEL, D_FF);
  transpose_cast_bf16<<<dim3(D_FF / 32, D_MODEL / 32), 256, 0, stream>>>(
      w_up, wup, D_MODEL, D_FF);
  transpose_cast_bf16<<<dim3(D_MODEL / 32, D_FF / 32), 256, 0, stream>>>(
      w_down, wdown, D_FF, D_MODEL);

  // 2) norm1 -> bf16
  rmsnorm_bf16<<<NTOK, 256, 0, stream>>>(x, norm1_w, h1);

  // 3) qkv = h1 @ qkv_w + qkv_b    (M=4096, N=6144, K=2048)
  gemm_bf16<true, false><<<dim3(3 * D_MODEL / 128, NTOK / 256), 256, 0, stream>>>(
      h1, wqkv, qkv_b, nullptr, qkvf, D_MODEL, 3 * D_MODEL);

  // 4) rope + layout
  rope_qk<<<(BB * TT * N_HEADS * (HEAD_DIM / 2)) / 256, 256, 0, stream>>>(qkvf, qb, kbuf);
  vcast_t<<<(BB * TT * D_MODEL) / 256, 256, 0, stream>>>(qkvf, vt);

  // 5) flash attention -> bf16 (B,T,D head-concat)
  flash_attn<<<dim3(TT / 128, BB * N_HEADS), 256, 0, stream>>>(qb, kbuf, vt, attnb);

  // 6) x1 = x + attn @ out_w + out_b  -> d_out
  gemm_bf16<true, true><<<dim3(D_MODEL / 128, NTOK / 256), 256, 0, stream>>>(
      attnb, wout, out_b, x, out, D_MODEL, D_MODEL);

  // 7) norm2 -> bf16
  rmsnorm_bf16<<<NTOK, 256, 0, stream>>>(out, norm2_w, h2);

  // 8) act = silu(h2@w_gate) * (h2@w_up)   (N=8192, K=2048)
  gemm_gateup<<<dim3(D_FF / 128, NTOK / 128), 256, 0, stream>>>(
      h2, wgate, wup, actb, D_MODEL, D_FF);

  // 9) out = x1 + act @ w_down   (K=8192, N=2048, in-place residual on d_out)
  gemm_bf16<false, true><<<dim3(D_MODEL / 128, NTOK / 256), 256, 0, stream>>>(
      actb, wdown, nullptr, out, out, D_FF, D_MODEL);
}
